// Devign1_33243046871386
// MI455X (gfx1250) — matrix operand order repaired and checked by reading, not compiled, not run on hardware
//
#include <hip/hip_runtime.h>
#include <math.h>

#define NN     50000
#define NEDGE  300000
#define INDIM  128
#define C      256
#define NLAYER 6
#define NGRAPH 256
#define KD     256   // K dimension of all GEMMs
#define G3     768   // 3*C

typedef __attribute__((ext_vector_type(16))) _Float16 v16h;
typedef __attribute__((ext_vector_type(8)))  float    v8f;
typedef _Float16 half_t;

// ---------------- utility kernels ----------------

__global__ void k_zero_f32(float* __restrict__ p, int n) {
    int i = blockIdx.x * blockDim.x + threadIdx.x;
    if (i < n) p[i] = 0.0f;
}

// H[n, 0:128] = x[n,:], H[n,128:256] = 0
__global__ void k_init_h(const float* __restrict__ x, float* __restrict__ H) {
    int i = blockIdx.x * blockDim.x + threadIdx.x;
    if (i >= NN * C) return;
    int n = i >> 8, c = i & 255;
    H[i] = (c < INDIM) ? x[n * INDIM + c] : 0.0f;
}

// direct f32 -> f16 convert (w_ih / w_hh are already [N,K] row-major = B^T)
__global__ void k_cvt_f16(const float* __restrict__ src, half_t* __restrict__ dst, int n) {
    int i = blockIdx.x * blockDim.x + threadIdx.x;
    if (i < n) dst[i] = (half_t)src[i];
}

// layer weights w[l][k][n] -> wt[l][n][k] f16  (B^T layout for WMMA B operand)
__global__ void k_transpose_cvt(const float* __restrict__ w, half_t* __restrict__ wt) {
    int i = blockIdx.x * blockDim.x + threadIdx.x;
    if (i >= NLAYER * C * C) return;
    int l = i >> 16;
    int r = i & 65535;
    int n = r >> 8;
    int k = r & 255;
    wt[(l << 16) + (n << 8) + k] = (half_t)w[(l << 16) + (k << 8) + n];
}

// ---------------- WMMA GEMM ----------------
// C[rows, cols] = A[rows, 256] (f32, cvt->f16) @ B[256, cols]
// Bt is B^T: [cols, 256] f16 row-major.  One wave -> 16x64 output tile.
__global__ __launch_bounds__(256) void k_gemm_wmma(
    const float*  __restrict__ A,
    const half_t* __restrict__ Bt,
    void*         __restrict__ Cout,
    int cols, int rows, int c_is_half)
{
    const int wave = blockIdx.x * (blockDim.x >> 5) + (threadIdx.x >> 5);
    const int lane = threadIdx.x & 31;
    const int colTiles = cols >> 6;          // 64-wide column groups
    const int rowTiles = rows >> 4;
    if (wave >= rowTiles * colTiles) return;
    const int rt = wave / colTiles;
    const int ct = wave % colTiles;

    const int hh = lane >> 4;                // 0 or 1 (half-wave)
    const int m  = lane & 15;

    const float*  arow  = A  + (size_t)(rt * 16 + m) * KD;
    const half_t* bbase = Bt + (size_t)(ct * 64 + m) * KD;

    v8f acc0 = {}, acc1 = {}, acc2 = {}, acc3 = {};

    for (int kk = 0; kk < KD; kk += 32) {
        // ---- A fragment: lane holds M=m, K = k0..k0+15 contiguous
        const int k0 = kk + 16 * hh;
        float4 f0 = *(const float4*)(arow + k0);
        float4 f1 = *(const float4*)(arow + k0 + 4);
        float4 f2 = *(const float4*)(arow + k0 + 8);
        float4 f3 = *(const float4*)(arow + k0 + 12);
        v16h a;
        a[0]  = (_Float16)f0.x; a[1]  = (_Float16)f0.y; a[2]  = (_Float16)f0.z; a[3]  = (_Float16)f0.w;
        a[4]  = (_Float16)f1.x; a[5]  = (_Float16)f1.y; a[6]  = (_Float16)f1.z; a[7]  = (_Float16)f1.w;
        a[8]  = (_Float16)f2.x; a[9]  = (_Float16)f2.y; a[10] = (_Float16)f2.z; a[11] = (_Float16)f2.w;
        a[12] = (_Float16)f3.x; a[13] = (_Float16)f3.y; a[14] = (_Float16)f3.z; a[15] = (_Float16)f3.w;

        // ---- B fragments: lane holds N = ct*64 + s*16 + m, K = kb..kb+15 contiguous
        const int kb = kk + 16 * hh;
        v16h b0 = *(const v16h*)(bbase + kb);
        v16h b1 = *(const v16h*)(bbase + 16 * KD + kb);
        v16h b2 = *(const v16h*)(bbase + 32 * KD + kb);
        v16h b3 = *(const v16h*)(bbase + 48 * KD + kb);

        acc0 = __builtin_amdgcn_wmma_f32_16x16x32_f16(false, a, false, b0, (short)0, acc0, false, false);
        acc1 = __builtin_amdgcn_wmma_f32_16x16x32_f16(false, a, false, b1, (short)0, acc1, false, false);
        acc2 = __builtin_amdgcn_wmma_f32_16x16x32_f16(false, a, false, b2, (short)0, acc2, false, false);
        acc3 = __builtin_amdgcn_wmma_f32_16x16x32_f16(false, a, false, b3, (short)0, acc3, false, false);
    }

    // ---- store: C/D layout -> VGPR i, lane l: M = i + 8*(l/16), N = l%16
    const int colBase = ct * 64 + m;
    if (c_is_half) {
        half_t* Cp = (half_t*)Cout;
        #pragma unroll
        for (int i = 0; i < 8; ++i) {
            size_t row = (size_t)(rt * 16 + i + 8 * hh) * cols + colBase;
            Cp[row +  0] = (half_t)acc0[i];
            Cp[row + 16] = (half_t)acc1[i];
            Cp[row + 32] = (half_t)acc2[i];
            Cp[row + 48] = (half_t)acc3[i];
        }
    } else {
        float* Cp = (float*)Cout;
        #pragma unroll
        for (int i = 0; i < 8; ++i) {
            size_t row = (size_t)(rt * 16 + i + 8 * hh) * cols + colBase;
            Cp[row +  0] = acc0[i];
            Cp[row + 16] = acc1[i];
            Cp[row + 32] = acc2[i];
            Cp[row + 48] = acc3[i];
        }
    }
}

// ---------------- edge scatter: AGG[dst] += M[src] ----------------
__global__ void k_scatter(const int* __restrict__ ei,
                          const float* __restrict__ M,
                          float* __restrict__ AGG)
{
    long long gid = (long long)blockIdx.x * blockDim.x + threadIdx.x;
    if (gid >= (long long)NEDGE * 64) return;
    int e   = (int)(gid >> 6);
    int seg = ((int)gid & 63) << 2;          // 4-float segment
    int src = ei[e];
    int dst = ei[NEDGE + e];
    float4 v = *(const float4*)(M + (size_t)src * C + seg);
    float* p = AGG + (size_t)dst * C + seg;
    atomicAdd(p + 0, v.x);
    atomicAdd(p + 1, v.y);
    atomicAdd(p + 2, v.z);
    atomicAdd(p + 3, v.w);
}

// ---------------- GRU cell update ----------------
__global__ void k_gru(const half_t* __restrict__ GI, const half_t* __restrict__ GH,
                      float* __restrict__ H,
                      const float* __restrict__ b_ih, const float* __restrict__ b_hh)
{
    int i = blockIdx.x * blockDim.x + threadIdx.x;
    if (i >= NN * C) return;
    int n = i >> 8, c = i & 255;
    const half_t* gi = GI + (size_t)n * G3;
    const half_t* gh = GH + (size_t)n * G3;
    float ir  = (float)gi[c]         + b_ih[c];
    float iz  = (float)gi[C + c]     + b_ih[C + c];
    float in_ = (float)gi[2 * C + c] + b_ih[2 * C + c];
    float hr  = (float)gh[c]         + b_hh[c];
    float hz  = (float)gh[C + c]     + b_hh[C + c];
    float hn  = (float)gh[2 * C + c] + b_hh[2 * C + c];
    float r   = 1.0f / (1.0f + __expf(-(ir + hr)));
    float z   = 1.0f / (1.0f + __expf(-(iz + hz)));
    float nn  = tanhf(in_ + r * hn);
    float h   = H[i];
    H[i] = (1.0f - z) * nn + z * h;
}

// ---------------- mean pool (sums + counts via L2-resident atomics) ----------------
__global__ void k_pool(const float* __restrict__ H, const int* __restrict__ batch,
                       float* __restrict__ gsum, float* __restrict__ gcnt)
{
    int i = blockIdx.x * blockDim.x + threadIdx.x;
    if (i >= NN * C) return;
    int n = i >> 8, c = i & 255;
    int b = batch[n];
    atomicAdd(&gsum[(size_t)b * C + c], H[i]);
    if (c == 0) atomicAdd(&gcnt[b], 1.0f);
}

// ---------------- classifier: relu(g@W1^T+b1)@W2^T+b2 -> sigmoid ----------------
__global__ __launch_bounds__(128) void k_cls(
    const float* __restrict__ gsum, const float* __restrict__ gcnt,
    const float* __restrict__ W1, const float* __restrict__ b1,
    const float* __restrict__ W2, const float* __restrict__ b2,
    float* __restrict__ out)
{
    int g = blockIdx.x;
    int j = threadIdx.x;                 // 0..127
    __shared__ float gm[C];
    __shared__ float red[128];
    float inv = 1.0f / fmaxf(gcnt[g], 1.0f);
    gm[j]       = gsum[(size_t)g * C + j] * inv;
    gm[j + 128] = gsum[(size_t)g * C + j + 128] * inv;
    __syncthreads();
    float acc = b1[j];
    #pragma unroll 8
    for (int k = 0; k < C; ++k) acc += gm[k] * W1[j * C + k];
    red[j] = fmaxf(acc, 0.0f) * W2[j];
    __syncthreads();
    for (int s = 64; s > 0; s >>= 1) {
        if (j < s) red[j] += red[j + s];
        __syncthreads();
    }
    if (j == 0) out[g] = 1.0f / (1.0f + __expf(-(red[0] + b2[0])));
}

// ---------------- host side ----------------

extern "C" void kernel_launch(void* const* d_in, const int* in_sizes, int n_in,
                              void* d_out, int out_size, void* d_ws, size_t ws_size,
                              hipStream_t stream) {
    const float* x    = (const float*)d_in[0];
    const int*   ei   = (const int*)  d_in[1];   // [2, NEDGE]
    const int*   bat  = (const int*)  d_in[2];   // [NN]
    const float* w    = (const float*)d_in[3];   // [6,256,256]
    const float* wih  = (const float*)d_in[4];   // [768,256]
    const float* whh  = (const float*)d_in[5];   // [768,256]
    const float* bih  = (const float*)d_in[6];
    const float* bhh  = (const float*)d_in[7];
    const float* W1   = (const float*)d_in[8];
    const float* b1   = (const float*)d_in[9];
    const float* W2   = (const float*)d_in[10];
    const float* b2   = (const float*)d_in[11];
    float* out = (float*)d_out;

    char* ws = (char*)d_ws;
    size_t off = 0;
    auto alloc = [&](size_t bytes) -> void* {
        void* p = ws + off;
        off += (bytes + 255) & ~(size_t)255;
        return p;
    };

    float*  H    = (float*) alloc((size_t)NN * C * 4);        // 51.2 MB
    float*  M    = (float*) alloc((size_t)NN * C * 4);        // 51.2 MB
    float*  AGG  = (float*) alloc((size_t)NN * C * 4);        // 51.2 MB
    half_t* GI   = (half_t*)alloc((size_t)NN * G3 * 2);       // 76.8 MB
    half_t* GH   = (half_t*)alloc((size_t)NN * G3 * 2);       // 76.8 MB
    half_t* WT   = (half_t*)alloc((size_t)NLAYER * C * C * 2);
    half_t* WIH  = (half_t*)alloc((size_t)G3 * C * 2);
    half_t* WHH  = (half_t*)alloc((size_t)G3 * C * 2);
    float*  GSUM = (float*) alloc((size_t)NGRAPH * C * 4);
    float*  GCNT = (float*) alloc((size_t)NGRAPH * 4);
    (void)ws_size; (void)n_in; (void)in_sizes; (void)out_size;

    const int TPB = 256;
    auto blk = [](long long n, int t) { return (int)((n + t - 1) / t); };

    // prep
    k_init_h<<<blk((long long)NN * C, TPB), TPB, 0, stream>>>(x, H);
    k_transpose_cvt<<<blk((long long)NLAYER * C * C, TPB), TPB, 0, stream>>>(w, WT);
    k_cvt_f16<<<blk((long long)G3 * C, TPB), TPB, 0, stream>>>(wih, WIH, G3 * C);
    k_cvt_f16<<<blk((long long)G3 * C, TPB), TPB, 0, stream>>>(whh, WHH, G3 * C);

    const int wavesPerBlk = TPB / 32;
    const int tilesC  = (NN / 16) * (C  / 64);   // 12500
    const int tilesG3 = (NN / 16) * (G3 / 64);   // 37500

    for (int l = 0; l < NLAYER; ++l) {
        // M = H @ W_l
        k_gemm_wmma<<<blk(tilesC, wavesPerBlk), TPB, 0, stream>>>(
            H, WT + (size_t)l * C * C, (void*)M, C, NN, 0);
        // AGG = scatter_add(M[src] -> dst)
        k_zero_f32<<<blk((long long)NN * C, TPB), TPB, 0, stream>>>(AGG, NN * C);
        k_scatter<<<blk((long long)NEDGE * 64, TPB), TPB, 0, stream>>>(ei, M, AGG);
        // GI = AGG @ w_ih^T ; GH = H @ w_hh^T   (biases folded into GRU kernel)
        k_gemm_wmma<<<blk(tilesG3, wavesPerBlk), TPB, 0, stream>>>(
            AGG, WIH, (void*)GI, G3, NN, 1);
        k_gemm_wmma<<<blk(tilesG3, wavesPerBlk), TPB, 0, stream>>>(
            H, WHH, (void*)GH, G3, NN, 1);
        // H = GRU(AGG-message, H)
        k_gru<<<blk((long long)NN * C, TPB), TPB, 0, stream>>>(GI, GH, H, bih, bhh);
    }

    // mean pool + classifier
    k_zero_f32<<<blk((long long)NGRAPH * C, TPB), TPB, 0, stream>>>(GSUM, NGRAPH * C);
    k_zero_f32<<<1, NGRAPH, 0, stream>>>(GCNT, NGRAPH);
    k_pool<<<blk((long long)NN * C, TPB), TPB, 0, stream>>>(H, bat, GSUM, GCNT);
    k_cls<<<NGRAPH, 128, 0, stream>>>(GSUM, GCNT, W1, b1, W2, b2, out);
}
